// FP8Linear_44899588112656
// MI455X (gfx1250) — 8-bit matrix operand order repaired and checked by reading, compiled, not run on hardware
//
#include <hip/hip_runtime.h>
#include <hip/hip_bf16.h>

typedef __attribute__((ext_vector_type(16))) int   v16i;
typedef __attribute__((ext_vector_type(8)))  float v8f;

#define K_DIM 4096
#define N_DIM 4096
#define FP8_MAX_V 448.0f


__global__ void init_ws_kernel(unsigned* __restrict__ ws) {
    if (threadIdx.x == 0 && blockIdx.x == 0) ws[0] = 0u;
}

__global__ __launch_bounds__(256)
void amax_kernel(const float4* __restrict__ x, unsigned* __restrict__ ws, int n4) {
    __shared__ float red[256];
    float m = 0.0f;
    for (int i = blockIdx.x * blockDim.x + threadIdx.x; i < n4;
         i += gridDim.x * blockDim.x) {
        float4 v = x[i];
        m = fmaxf(m, fmaxf(fmaxf(fabsf(v.x), fabsf(v.y)),
                           fmaxf(fabsf(v.z), fabsf(v.w))));
    }
    red[threadIdx.x] = m;
    __syncthreads();
    for (int s = 128; s > 0; s >>= 1) {
        if ((int)threadIdx.x < s)
            red[threadIdx.x] = fmaxf(red[threadIdx.x], red[threadIdx.x + s]);
        __syncthreads();
    }
    if (threadIdx.x == 0) atomicMax(ws, __float_as_uint(red[0]));
}

#if defined(__has_builtin)
#if __has_builtin(__builtin_amdgcn_cvt_pk_fp8_f32)
#define HAVE_CVT_PK_FP8 1
#endif
#endif

#ifndef HAVE_CVT_PK_FP8
__device__ inline unsigned fp8_sw_byte(float f) {
    unsigned sign = (__float_as_uint(f) >> 24) & 0x80u;
    float ax = fminf(fabsf(f), FP8_MAX_V);
    unsigned code;
    if (ax < 0.015625f) {
        code = (unsigned)__float2int_rn(ax * 512.0f);
    } else {
        int e; float fr = frexpf(ax, &e);
        int E = e - 1 + 7;
        int mant = __float2int_rn((fr * 2.0f - 1.0f) * 8.0f);
        if (mant == 8) { mant = 0; E += 1; }
        code = (unsigned)((E << 3) | mant);
        if (code > 0x7Eu) code = 0x7Eu;
    }
    return sign | code;
}
#endif

__device__ inline unsigned pack4_fp8(float a, float b, float c, float d) {
#ifdef HAVE_CVT_PK_FP8
    int p = 0;
    p = __builtin_amdgcn_cvt_pk_fp8_f32(a, b, p, false);
    p = __builtin_amdgcn_cvt_pk_fp8_f32(c, d, p, true);
    return (unsigned)p;
#else
    return fp8_sw_byte(a) | (fp8_sw_byte(b) << 8) |
           (fp8_sw_byte(c) << 16) | (fp8_sw_byte(d) << 24);
#endif
}

__global__ __launch_bounds__(256)
void quant_kernel(const float4* __restrict__ x, const unsigned* __restrict__ ws,
                  uint2* __restrict__ xq, int n8) {
    int i = blockIdx.x * blockDim.x + threadIdx.x;
    if (i >= n8) return;
    float amax = __uint_as_float(ws[0]);
    float xscale = fmaxf(amax / FP8_MAX_V, 1e-12f);
    float inv = 1.0f / xscale;
    float4 a = x[2 * i + 0];
    float4 b = x[2 * i + 1];
    a.x = fminf(fmaxf(a.x * inv, -FP8_MAX_V), FP8_MAX_V);
    a.y = fminf(fmaxf(a.y * inv, -FP8_MAX_V), FP8_MAX_V);
    a.z = fminf(fmaxf(a.z * inv, -FP8_MAX_V), FP8_MAX_V);
    a.w = fminf(fmaxf(a.w * inv, -FP8_MAX_V), FP8_MAX_V);
    b.x = fminf(fmaxf(b.x * inv, -FP8_MAX_V), FP8_MAX_V);
    b.y = fminf(fmaxf(b.y * inv, -FP8_MAX_V), FP8_MAX_V);
    b.z = fminf(fmaxf(b.z * inv, -FP8_MAX_V), FP8_MAX_V);
    b.w = fminf(fmaxf(b.w * inv, -FP8_MAX_V), FP8_MAX_V);
    uint2 o;
    o.x = pack4_fp8(a.x, a.y, a.z, a.w);
    o.y = pack4_fp8(b.x, b.y, b.z, b.w);
    xq[i] = o;
}

union AFrag { v16i v; long d[8]; };
union BFrag { v16i v; int4 q[4]; };

__global__ __launch_bounds__(256)
void gemm_fp8_kernel(const unsigned char* __restrict__ xq,
                     const unsigned char* __restrict__ w,
                     const float* __restrict__ wscale_p,
                     const __hip_bfloat16* __restrict__ bias,
                     const unsigned* __restrict__ ws_amax,
                     __hip_bfloat16* __restrict__ out, int M) {
    const int lane  = threadIdx.x & 31;
    const int wave  = threadIdx.x >> 5;
    const int l     = lane & 15;
    const int hi    = (lane >> 4) & 1;
    const int waveM = wave & 3;
    const int waveN = wave >> 2;
    const int m0 = blockIdx.y * 128 + waveM * 32;
    const int n0 = blockIdx.x * 128 + waveN * 64;

    v8f z = {0.f, 0.f, 0.f, 0.f, 0.f, 0.f, 0.f, 0.f};
    v8f acc[2][4];
#pragma unroll
    for (int i = 0; i < 2; ++i)
#pragma unroll
        for (int j = 0; j < 4; ++j) acc[i][j] = z;

    const unsigned char* aRow[2];
#pragma unroll
    for (int i = 0; i < 2; ++i)
        aRow[i] = xq + (size_t)(m0 + i * 16 + l) * K_DIM + hi * 16;
    const unsigned char* bCol[4];
#pragma unroll
    for (int j = 0; j < 4; ++j)
        bCol[j] = w + (size_t)(n0 + j * 16 + l) * K_DIM + hi * 16;

    for (int k0 = 0; k0 < K_DIM; k0 += 128) {
        AFrag a[2]; BFrag b[4];
#pragma unroll
        for (int i = 0; i < 2; ++i)
#pragma unroll
            for (int c = 0; c < 4; ++c) {
                a[i].d[c]     = *(const long*)(aRow[i] + k0 + (c / 2) * 32 + (c % 2) * 8);
                a[i].d[4 + c] = *(const long*)(aRow[i] + k0 + 64 + (c / 2) * 32 + (c % 2) * 8);
            }
#pragma unroll
        for (int j = 0; j < 4; ++j)
#pragma unroll
            for (int c = 0; c < 4; ++c)
                b[j].q[c] = *(const int4*)(bCol[j] + k0 + c * 32);

#pragma unroll
        for (int i = 0; i < 2; ++i)
#pragma unroll
            for (int j = 0; j < 4; ++j)
                acc[i][j] = __builtin_amdgcn_wmma_f32_16x16x128_fp8_fp8(
                    a[i].v, b[j].v, (short)0, acc[i][j], false, false);
    }

    float amax  = __uint_as_float(ws_amax[0]);
    float xs    = fmaxf(amax / FP8_MAX_V, 1e-12f);
    float scale = xs * wscale_p[0];

#pragma unroll
    for (int i = 0; i < 2; ++i) {
#pragma unroll
        for (int j = 0; j < 4; ++j) {
            int n = n0 + j * 16 + l;
            float bv = __bfloat162float(bias[n]);
#pragma unroll
            for (int r = 0; r < 8; ++r) {
                int m = m0 + i * 16 + hi * 8 + r;
                float v = acc[i][j][r] * scale;
                float vr = __bfloat162float(__float2bfloat16(v));
                out[(size_t)m * N_DIM + n] = __float2bfloat16(vr + bv);
            }
        }
    }
}

extern "C" void kernel_launch(void* const* d_in, const int* in_sizes, int n_in,
                              void* d_out, int out_size, void* d_ws, size_t ws_size,
                              hipStream_t stream) {
    const float*           x      = (const float*)d_in[0];
    const unsigned char*   w_fp8  = (const unsigned char*)d_in[1];
    const float*           wscale = (const float*)d_in[2];
    const __hip_bfloat16*  bias   = (const __hip_bfloat16*)d_in[3];
    __hip_bfloat16*        out    = (__hip_bfloat16*)d_out;

    const int M  = in_sizes[0] / K_DIM;
    const int n4 = in_sizes[0] / 4;
    const int n8 = in_sizes[0] / 8;

    unsigned*      ws_amax = (unsigned*)d_ws;
    unsigned char* xq      = (unsigned char*)d_ws + 256;

    init_ws_kernel<<<1, 32, 0, stream>>>(ws_amax);
    amax_kernel<<<2048, 256, 0, stream>>>((const float4*)x, ws_amax, n4);
    quant_kernel<<<(n8 + 255) / 256, 256, 0, stream>>>(
        (const float4*)x, ws_amax, (uint2*)xq, n8);

    dim3 grid(N_DIM / 128, M / 128);
    gemm_fp8_kernel<<<grid, 256, 0, stream>>>(xq, w_fp8, wscale, bias,
                                              ws_amax, out, M);
}
